// SCSSD_64372969832967
// MI455X (gfx1250) — compile-verified, matrix operand order checked by reading, not run on hardware
//
#include <hip/hip_runtime.h>
#include <hip/hip_bf16.h>
#include <stdint.h>

// ---------------------------------------------------------------------------
// CDNA5 (gfx1250) implementation. wave32. WMMA f32<-f16 16x16x32 with all
// operands pre-packed into the ISA fragment layout (two b128 loads per
// fragment); fused per-sequence Mamba-2 scan kernels using big per-WGP LDS,
// with async global->LDS double buffering of the next chunk (ASYNCcnt path).
// ---------------------------------------------------------------------------

typedef __attribute__((ext_vector_type(16))) _Float16 v16h;
typedef __attribute__((ext_vector_type(8)))  float    v8f;

#define DEV __device__ __forceinline__

// K index inside a 16x16x32 f16 fragment for element e (0..15) of this lane.
// Per ISA 7.12.2: VGPR v pairs: v<4 -> K=2v,2v+1 ; v>=4 -> K=16+2(v-4),...
// lanes 16..31 add +8.
DEV int frag_k(int e, int lane) {
  int v = e >> 1, s = e & 1;
  int k = ((v < 4) ? (v << 1) : (16 + ((v - 4) << 1))) + s;
  return k + ((lane >> 4) << 3);
}

// Inverse map: given k (0..31) -> lane-half and element index.
DEV int inv_lane_hi(int k) { return (k >> 3) & 1; }
DEV int inv_elem(int k)    { return (k & 7) | ((k >> 4) << 3); }

// A fragment gathered from f32 memory: element (m,k) at A[m*rs + k*cs].
DEV v16h frag_a(const float* A, int rs, int cs, int kmax, int lane) {
  v16h f;
  int m = lane & 15;
#pragma unroll
  for (int e = 0; e < 16; ++e) {
    int k = frag_k(e, lane);
    f[e] = (k < kmax) ? (_Float16)A[m * rs + k * cs] : (_Float16)0.0f;
  }
  return f;
}

DEV v8f wmma32(v16h a, v16h b, v8f c) {
  return __builtin_amdgcn_wmma_f32_16x16x32_f16(false, a, false, b, (short)0, c,
                                                false, false);
}

DEV float siluf(float v)     { return v / (1.0f + expf(-v)); }
DEV float softplusf(float v) { return (v > 20.0f) ? v : log1pf(expf(v)); }
DEV float geluf(float v) {
  float t = 0.7978845608028654f * (v + 0.044715f * v * v * v);
  return 0.5f * v * (1.0f + tanhf(t));
}

// ---------------------------------------------------------------------------
// Pack a row-major (N x K) f32 weight into fragment-order f16.
__global__ __launch_bounds__(256) void pack_w_kernel(
    const float* __restrict__ w, _Float16* __restrict__ dst, int rowlen,
    int NT, int KT, int Kmax, int Nmax, int total) {
  int i = blockIdx.x * 256 + threadIdx.x;
  if (i >= total) return;
  int e = i & 15, lane = (i >> 4) & 31, f = i >> 9;
  int nt = f % NT, kt = f / NT;
  int n = nt * 16 + (lane & 15);
  int k = kt * 32 + frag_k(e, lane);
  dst[i] = (k < Kmax && n < Nmax) ? (_Float16)w[(size_t)n * rowlen + k]
                                  : (_Float16)0.0f;
}

// ---------------------------------------------------------------------------
// Fused sequential Mamba-2 (chunked SSD, chunk=64, headdim=8) - one workgroup
// per sequence; state + chunk workspace resident in LDS across the scan.
// ---------------------------------------------------------------------------
template <int DMODEL, int DIN, int NH, int DSTATE>
constexpr size_t mamba_lds_floats() {
  constexpr int DI = NH * 8, DCONV = DI + 2 * DSTATE;
  constexpr int KT = (DSTATE + 31) / 32;
  return (size_t)2 * 64 * DMODEL +  // async x stage (double buffer)
         (size_t)64 * DIN + 2 * DCONV + 64 * NH + 3 * (NH * 64) + 8 * 256 +
         64 * NH * 8 + NH * 8 * DSTATE + 64 * DI + 64 * 4 + 2048 * KT;
}

template <int DMODEL, int DIN, int NH, int DSTATE>
__global__ __launch_bounds__(256) void mamba_seq_kernel(
    const float* __restrict__ xin, float* __restrict__ yout,
    const float* __restrict__ resid, const float* __restrict__ resid_scale,
    const _Float16* __restrict__ in_pk, const float* __restrict__ in_b,
    const float* __restrict__ conv_w, const float* __restrict__ conv_b,
    const float* __restrict__ A_log, const float* __restrict__ Dp,
    const float* __restrict__ dt_bias, const float* __restrict__ rms_w,
    const float* __restrict__ out_w, int L) {
  constexpr int DI = NH * 8;
  constexpr int DCONV = DI + 2 * DSTATE;
  constexpr int NT = (DIN + 15) / 16;
  constexpr int KT = (DSTATE + 31) / 32;
  constexpr int CREG = (64 * DCONV + 255) / 256;
  constexpr int CPB = 64 * DMODEL * 4;  // bytes per x chunk

  extern __shared__ float smem[];
  float* s_stage = smem;                    // [2][64*DMODEL] async x stage
  float* s_zx   = s_stage + 2 * 64 * DMODEL;  // [64][DIN]  z|xBC|dt chunk
  float* s_hist = s_zx + 64 * DIN;         // [2][DCONV] raw conv history
  float* s_dt   = s_hist + 2 * DCONV;      // [64][NH]
  float* s_acs  = s_dt + 64 * NH;          // [NH][64]   cumsum(dt*A)
  float* s_dec  = s_acs + NH * 64;         // [NH][64]
  float* s_eac  = s_dec + NH * 64;         // [NH][64]
  float* s_tile = s_eac + NH * 64;         // [8][16][16] per-wave masked CB
  float* s_yd   = s_tile + 8 * 256;        // [64][NH][8]
  float* s_st   = s_yd + 64 * NH * 8;      // [NH][8][DSTATE]
  float* s_yn   = s_st + NH * 8 * DSTATE;  // [64][DI]
  float* s_red  = s_yn + 64 * DI;          // [64][4]
  _Float16* s_apk = (_Float16*)(s_red + 64 * 4);  // Cm A-frags [4][KT][32][16]
  _Float16* s_bpk = s_apk + 4 * KT * 512;         // Bm B-frags [4][KT][32][16]

  const int tid  = threadIdx.x;
  const int lane = tid & 31;
  const int wave = tid >> 5;
  const size_t sbase = (size_t)blockIdx.x * L * DMODEL;
  const float* xs_in = xin + sbase;
  float* ys_out = yout + sbase;
  const float rscale = resid_scale ? *resid_scale : 1.0f;
  const v16h* in_pkv = (const v16h*)in_pk;

  // Async copy of one x chunk (contiguous CPB bytes) into a stage buffer:
  // each participating lane moves 16B via GLOBAL_LOAD_ASYNC_TO_LDS_B128
  // (tracked by ASYNCcnt; waited with s_wait_asynccnt at the next chunk top).
  auto issue_chunk_copy = [&](int cn, int buf) {
    if (tid * 16 < CPB) {
      unsigned int ldsoff =
          (unsigned int)(uintptr_t)(s_stage + buf * 64 * DMODEL) +
          (unsigned int)(tid * 16);
      unsigned long long ga =
          (unsigned long long)(uintptr_t)(xs_in + (size_t)(cn << 6) * DMODEL) +
          (unsigned long long)(tid * 16);
      asm volatile("global_load_async_to_lds_b128 %0, %1, off"
                   :: "v"(ldsoff), "v"(ga)
                   : "memory");
    }
  };

  float Aneg[NH];
#pragma unroll
  for (int h = 0; h < NH; ++h) Aneg[h] = -expf(A_log[h]);

  for (int e = tid; e < 2 * DCONV; e += 256) s_hist[e] = 0.0f;
  for (int e = tid; e < NH * 8 * DSTATE; e += 256) s_st[e] = 0.0f;
  issue_chunk_copy(0, 0);
  __syncthreads();

  const int nchunks = L >> 6;
  for (int c = 0; c < nchunks; ++c) {
    const int buf = c & 1;
    asm volatile("s_wait_asynccnt 0" ::: "memory");  // my stage writes landed
    __syncthreads();                                 // everyone's landed
    if (c + 1 < nchunks) issue_chunk_copy(c + 1, buf ^ 1);
    const float* xc = s_stage + buf * 64 * DMODEL;   // LDS-resident x chunk

    // ---- Phase 1: zxbcdt = x @ in_w^T + in_b  (WMMA, packed weights) ------
    for (int t = wave; t < 4 * NT; t += 8) {
      int nt = t >> 2, mt = t & 3;
      int nmax = DIN - nt * 16; if (nmax > 16) nmax = 16;
      v8f acc = {};
      v16h af = frag_a(xc + (size_t)(mt * 16) * DMODEL, DMODEL, 1, DMODEL, lane);
      v16h bf = in_pkv[nt * 32 + lane];
      acc = wmma32(af, bf, acc);
      int n = lane & 15;
      int hi = (lane >> 4) << 3;
      if (n < nmax) {
        float bias = in_b[nt * 16 + n];
#pragma unroll
        for (int v = 0; v < 8; ++v)
          s_zx[(mt * 16 + v + hi) * DIN + nt * 16 + n] = acc[v] + bias;
      }
    }
    __syncthreads();

    // ---- Phase 2: causal depthwise conv(K=3) + SiLU, in place -------------
    float creg[CREG];
    {
      int i = 0;
      for (int e = tid; e < 64 * DCONV; e += 256, ++i) {
        int l = e / DCONV, ch = e - l * DCONV;
        float xm2 = (l >= 2) ? s_zx[(l - 2) * DIN + DI + ch] : s_hist[l * DCONV + ch];
        float xm1 = (l >= 1) ? s_zx[(l - 1) * DIN + DI + ch] : s_hist[DCONV + ch];
        float x0  = s_zx[l * DIN + DI + ch];
        float v = conv_w[ch * 3 + 0] * xm2 + conv_w[ch * 3 + 1] * xm1 +
                  conv_w[ch * 3 + 2] * x0 + conv_b[ch];
        creg[i] = siluf(v);
      }
    }
    __syncthreads();
    for (int e = tid; e < 2 * DCONV; e += 256) {
      int l = e / DCONV, ch = e - l * DCONV;
      s_hist[e] = s_zx[(62 + l) * DIN + DI + ch];  // raw tail for next chunk
    }
    __syncthreads();
    {
      int i = 0;
      for (int e = tid; e < 64 * DCONV; e += 256, ++i) {
        int l = e / DCONV, ch = e - l * DCONV;
        s_zx[l * DIN + DI + ch] = creg[i];
      }
    }
    __syncthreads();

    // ---- Phase 3: dt = softplus(.+bias); acs = cumsum(dt*A); decays -------
    for (int e = tid; e < 64 * NH; e += 256) {
      int l = e / NH, h = e - l * NH;
      float dt = softplusf(s_zx[l * DIN + (DIN - NH) + h] + dt_bias[h]);
      s_dt[l * NH + h] = dt;
      s_acs[h * 64 + l] = dt * Aneg[h];
    }
    __syncthreads();
    if (tid < NH) {
      float run = 0.0f;
      for (int l = 0; l < 64; ++l) { run += s_acs[tid * 64 + l]; s_acs[tid * 64 + l] = run; }
    }
    __syncthreads();
    for (int e = tid; e < NH * 64; e += 256) {
      int h = e >> 6;
      s_dec[e] = expf(s_acs[h * 64 + 63] - s_acs[e]);
      s_eac[e] = expf(s_acs[e]);
    }
    __syncthreads();

    // ---- Phase 4a: pack Cm (A-frags) / Bm (B-frags) to f16 in LDS ---------
    for (int e2 = tid; e2 < 4 * KT * 512; e2 += 256) {
      int e = e2 & 15, ln = (e2 >> 4) & 31, f = e2 >> 9;
      int kt = f % KT, tt = f / KT;
      int k = kt * 32 + frag_k(e, ln);
      int row = tt * 16 + (ln & 15);
      s_apk[e2] = (k < DSTATE) ? (_Float16)s_zx[row * DIN + 2 * DI + DSTATE + k]
                               : (_Float16)0.0f;
      s_bpk[e2] = (k < DSTATE) ? (_Float16)s_zx[row * DIN + 2 * DI + k]
                               : (_Float16)0.0f;
    }

    // ---- Phase 4b: Yo from previous inter-chunk state ---------------------
    for (int e = tid; e < 64 * NH * 8; e += 256) {
      int p = e & 7, t2 = e >> 3;
      int h = t2 % NH, l = t2 / NH;
      const float* Cm = &s_zx[l * DIN + 2 * DI + DSTATE];
      const float* st = &s_st[(h * 8 + p) * DSTATE];
      float acc = 0.0f;
      for (int n = 0; n < DSTATE; ++n) acc += Cm[n] * st[n];
      s_yd[(l * NH + h) * 8 + p] = s_eac[h * 64 + l] * acc;
    }
    __syncthreads();

    // ---- Phase 5: Yd intra-chunk: CB = Cm@Bm^T via WMMA, mask, apply x ----
    {
      const v16h* apk = (const v16h*)s_apk;
      const v16h* bpk = (const v16h*)s_bpk;
      for (int item = wave; item < NH * 4; item += 8) {
        int h = item >> 2, lt = item & 3;  // wave owns (h, l-tile) rows
        float* tile = &s_tile[wave * 256];
        for (int st = 0; st <= lt; ++st) {
          v8f acc = {};
#pragma unroll
          for (int kt = 0; kt < KT; ++kt)
            acc = wmma32(apk[(lt * KT + kt) * 32 + lane],
                         bpk[(st * KT + kt) * 32 + lane], acc);
          int n = lane & 15;
          int hi = (lane >> 4) << 3;
#pragma unroll
          for (int v = 0; v < 8; ++v) {
            int lr = lt * 16 + v + hi, sc = st * 16 + n;
            float m = (sc <= lr) ? expf(s_acs[h * 64 + lr] - s_acs[h * 64 + sc]) : 0.0f;
            tile[(v + hi) * 16 + n] = acc[v] * m;  // same-wave LDS, DS in-order
          }
#pragma unroll
          for (int i = 0; i < 4; ++i) {
            int idx = lane * 4 + i, lp = idx >> 3, p = idx & 7;
            float sum = 0.0f;
            for (int sp = 0; sp < 16; ++sp) {
              int sa = st * 16 + sp;
              float xv = s_zx[sa * DIN + DI + h * 8 + p] * s_dt[sa * NH + h];
              sum += tile[lp * 16 + sp] * xv;
            }
            s_yd[((lt * 16 + lp) * NH + h) * 8 + p] += sum;  // wave-exclusive
          }
        }
      }
    }
    __syncthreads();

    // ---- Phase 6: state = state*exp(acs[63]) + Bm^T(dec*x) ----------------
    for (int e = tid; e < NH * 8 * DSTATE; e += 256) {
      int n = e % DSTATE, t2 = e / DSTATE;
      int p = t2 & 7, h = t2 >> 3;
      float acc = s_st[(h * 8 + p) * DSTATE + n] * expf(s_acs[h * 64 + 63]);
      for (int l = 0; l < 64; ++l) {
        float xv = s_zx[l * DIN + DI + h * 8 + p] * s_dt[l * NH + h];
        acc += s_zx[l * DIN + 2 * DI + n] * s_dec[h * 64 + l] * xv;
      }
      s_st[(h * 8 + p) * DSTATE + n] = acc;
    }
    __syncthreads();

    // ---- Phase 7: +D*xs, gate silu(z), RMS-norm, out-proj, +resid ---------
    for (int e = tid; e < 64 * DI; e += 256) {
      int l = e / DI, cc = e - l * DI;
      int h = cc >> 3, p = cc & 7;
      float xsv = s_zx[l * DIN + DI + cc];
      float yv = s_yd[(l * NH + h) * 8 + p] + Dp[h] * xsv;
      s_yn[l * DI + cc] = yv * siluf(s_zx[l * DIN + cc]);
    }
    __syncthreads();
    for (int e = tid; e < 64 * 4; e += 256) {
      int l = e >> 2, q = e & 3;
      float ss = 0.0f;
      for (int j = q; j < DI; j += 4) { float v = s_yn[l * DI + j]; ss += v * v; }
      s_red[e] = ss;
    }
    __syncthreads();
    for (int e = tid; e < 64 * DMODEL; e += 256) {
      int l = e / DMODEL, m = e - l * DMODEL;
      float ms = (s_red[l * 4] + s_red[l * 4 + 1] + s_red[l * 4 + 2] + s_red[l * 4 + 3]) / (float)DI;
      float sc = rsqrtf(ms + 1e-5f);
      float acc = 0.0f;
      for (int cc = 0; cc < DI; ++cc)
        acc += (s_yn[l * DI + cc] * sc * rms_w[cc]) * out_w[m * DI + cc];
      size_t gi = (size_t)((c << 6) + l) * DMODEL + m;
      if (resid) acc += rscale * resid[sbase + gi];
      ys_out[gi] = acc;
    }
    __syncthreads();
  }
}

// ---------------------------------------------------------------------------
// gf[b,l,g] = sum_c x[b,c,l] * tokw[g,c] + tokb[g]
__global__ __launch_bounds__(256) void seed_tok_kernel(
    const float* __restrict__ x, const float* __restrict__ tw,
    const float* __restrict__ tb, float* __restrict__ gf) {
  int i = blockIdx.x * 256 + threadIdx.x;
  if (i >= 4 * 4096 * 8) return;
  int g = i & 7, l = (i >> 3) & 4095, b = i >> 15;
  const float* xb = x + (size_t)b * 128 * 4096 + l;
  const float* w = tw + g * 128;
  float acc = tb[g];
  for (int cc = 0; cc < 128; ++cc) acc += xb[(size_t)cc * 4096] * w[cc];
  gf[i] = acc;
}

// seed[b,c,l] = x[b,c,l] + gf2[b,l,:] . detok_w[c,:] + detok_b[c]
__global__ __launch_bounds__(256) void seed_combine_kernel(
    const float* __restrict__ x, const float* __restrict__ gf2,
    const float* __restrict__ dw, const float* __restrict__ db,
    float* __restrict__ seed) {
  int i = blockIdx.x * 256 + threadIdx.x;
  if (i >= 4 * 128 * 4096) return;
  int l = i & 4095, cc = (i >> 12) & 127, b = i >> 19;
  const float* gfp = gf2 + ((size_t)b * 4096 + l) * 8;
  float acc = db[cc];
#pragma unroll
  for (int g = 0; g < 8; ++g) acc += gfp[g] * dw[cc * 8 + g];
  seed[i] = x[i] + acc;
}

// LayerNorm over C=128 at each (b,l): xn[b,l,c]
__global__ __launch_bounds__(128) void norm1_kernel(
    const float* __restrict__ x, const float* __restrict__ w,
    const float* __restrict__ b, float* __restrict__ xn) {
  __shared__ float sb[128];
  int bl = blockIdx.x, bb = bl >> 12, l = bl & 4095, c = threadIdx.x;
  float v = x[((size_t)bb * 128 + c) * 4096 + l];
  sb[c] = v; __syncthreads();
  for (int s = 64; s > 0; s >>= 1) { if (c < s) sb[c] += sb[c + s]; __syncthreads(); }
  float mu = sb[0] / 128.0f; __syncthreads();
  float d = v - mu;
  sb[c] = d * d; __syncthreads();
  for (int s = 64; s > 0; s >>= 1) { if (c < s) sb[c] += sb[c + s]; __syncthreads(); }
  float var = sb[0] / 128.0f;
  xn[(size_t)bl * 128 + c] = d * rsqrtf(var + 1e-5f) * w[c] + b[c];
}

// xsplit[g,b,l,j] = xn[b,l,16g+j]
__global__ __launch_bounds__(256) void split_kernel(
    const float* __restrict__ xn, float* __restrict__ xsplit) {
  int i = blockIdx.x * 256 + threadIdx.x;
  if (i >= 8 * 4 * 4096 * 16) return;
  int j = i & 15, l = (i >> 4) & 4095, b = (i >> 16) & 3, g = i >> 18;
  xsplit[i] = xn[((size_t)b * 4096 + l) * 128 + g * 16 + j];
}

// gm[g*4+b, j] = mean_l ysB[g,b,l,j]
__global__ __launch_bounds__(256) void group_mean_kernel(
    const float* __restrict__ ysB, float* __restrict__ gm) {
  __shared__ float sb[256];
  int seq = blockIdx.x;
  float part[16];
#pragma unroll
  for (int j = 0; j < 16; ++j) part[j] = 0.0f;
  for (int l = threadIdx.x; l < 4096; l += 256) {
    const float* p = ysB + ((size_t)seq * 4096 + l) * 16;
#pragma unroll
    for (int j = 0; j < 16; ++j) part[j] += p[j];
  }
  for (int j = 0; j < 16; ++j) {
    sb[threadIdx.x] = part[j]; __syncthreads();
    for (int s = 128; s > 0; s >>= 1) {
      if (threadIdx.x < s) sb[threadIdx.x] += sb[threadIdx.x + s];
      __syncthreads();
    }
    if (threadIdx.x == 0) gm[seq * 16 + j] = sb[0] / 4096.0f;
    __syncthreads();
  }
}

// ys + softmax(crossW)-mixed means -> LN(norm2) -> + seed -> out (B,C,L)
// Also emits the fc1 A operand as fragment-packed f16.
__global__ __launch_bounds__(128) void mix_norm2_kernel(
    const float* __restrict__ ysB, const float* __restrict__ gm,
    const float* __restrict__ crossW, const float* __restrict__ seed,
    const float* __restrict__ n2w, const float* __restrict__ n2b,
    float* __restrict__ outb, _Float16* __restrict__ apack1) {
  __shared__ float sb[128];
  int bl = blockIdx.x, bb = bl >> 12, l = bl & 4095, c = threadIdx.x;
  int g = c >> 4, j = c & 15;
  float val = ysB[((size_t)(g * 4 + bb) * 4096 + l) * 16 + j];
  float gmix = 0.0f;
  for (int q = 0; q < 8; ++q) {
    float mx = crossW[q * 8];
    for (int p = 1; p < 8; ++p) mx = fmaxf(mx, crossW[q * 8 + p]);
    float den = 0.0f, num = 0.0f;
    for (int p = 0; p < 8; ++p) {
      float e = expf(crossW[q * 8 + p] - mx);
      den += e; if (p == g) num = e;
    }
    gmix += gm[(q * 4 + bb) * 16 + j] * (num / den);
  }
  val += gmix;
  sb[c] = val; __syncthreads();
  for (int s = 64; s > 0; s >>= 1) { if (c < s) sb[c] += sb[c + s]; __syncthreads(); }
  float mu = sb[0] / 128.0f; __syncthreads();
  float d = val - mu;
  sb[c] = d * d; __syncthreads();
  for (int s = 64; s > 0; s >>= 1) { if (c < s) sb[c] += sb[c + s]; __syncthreads(); }
  float var = sb[0] / 128.0f;
  size_t oi = ((size_t)bb * 128 + c) * 4096 + l;
  float outv = d * rsqrtf(var + 1e-5f) * n2w[c] + n2b[c] + seed[oi];
  outb[oi] = outv;
  // packed A fragment element for fc1: m=l&15, k=c
  int mt = l >> 4, m = l & 15;
  int kt = c >> 5, kk = c & 31;
  int ln = m + (inv_lane_hi(kk) << 4);
  int e2 = inv_elem(kk);
  apack1[(((size_t)bb * 256 + mt) * 4 + kt) * 512 + ln * 16 + e2] = (_Float16)outv;
}

// h1[b,o,l] = out . fc1_w + b  (M=4096,N=256,K=128; all operands packed f16)
__global__ __launch_bounds__(256) void fc1_wmma_kernel(
    const _Float16* __restrict__ apack, const _Float16* __restrict__ wpack,
    const float* __restrict__ bias, float* __restrict__ h1) {
  int lane = threadIdx.x & 31, wave = threadIdx.x >> 5;
  int mt = blockIdx.x, nt = blockIdx.y * 8 + wave, b = blockIdx.z;
  const v16h* ap = (const v16h*)(apack + ((size_t)b * 256 + mt) * 4 * 512);
  const v16h* wp = (const v16h*)wpack;
  v8f acc = {};
#pragma unroll
  for (int kt = 0; kt < 4; ++kt) {
    if (kt < 3) __builtin_prefetch(&ap[(kt + 1) * 32 + lane], 0, 3);
    acc = wmma32(ap[kt * 32 + lane], wp[(kt * 16 + nt) * 32 + lane], acc);
  }
  int n = nt * 16 + (lane & 15);
  int hi = (lane >> 4) << 3;
  float bv = bias[n];
  float* p = h1 + ((size_t)b * 256 + n) * 4096 + mt * 16 + hi;
  *(float4*)p = make_float4(acc[0] + bv, acc[1] + bv, acc[2] + bv, acc[3] + bv);
  *(float4*)(p + 4) = make_float4(acc[4] + bv, acc[5] + bv, acc[6] + bv, acc[7] + bv);
}

// depthwise 3x3 SAME + tanh-GELU; writes fc2 A operand directly as packed f16
__global__ __launch_bounds__(256) void dwconv_gelu_kernel(
    const float* __restrict__ h1, const float* __restrict__ w,
    const float* __restrict__ bias, _Float16* __restrict__ apack2) {
  int i = blockIdx.x * 256 + threadIdx.x;
  if (i >= 4 * 256 * 4096) return;
  int ww = i & 63, hh = (i >> 6) & 63, o = (i >> 12) & 255, b = i >> 20;
  const float* base = h1 + ((size_t)b * 256 + o) * 4096;
  float acc = bias[o];
#pragma unroll
  for (int kh = 0; kh < 3; ++kh) {
    int y = hh + kh - 1;
    if (y < 0 || y > 63) continue;
#pragma unroll
    for (int kw = 0; kw < 3; ++kw) {
      int xx = ww + kw - 1;
      if (xx < 0 || xx > 63) continue;
      acc += base[y * 64 + xx] * w[o * 9 + kh * 3 + kw];
    }
  }
  int l = i & 4095;
  int mt = l >> 4, m = l & 15;
  int kt = o >> 5, kk = o & 31;
  int ln = m + (inv_lane_hi(kk) << 4);
  int e2 = inv_elem(kk);
  apack2[(((size_t)b * 256 + mt) * 8 + kt) * 512 + ln * 16 + e2] =
      (_Float16)geluf(acc);
}

// dout = out + h2 . fc2_w + b  (M=4096,N=128,K=256; packed f16 operands)
__global__ __launch_bounds__(256) void fc2_wmma_kernel(
    const _Float16* __restrict__ apack, const _Float16* __restrict__ wpack,
    const float* __restrict__ bias, const float* __restrict__ outb,
    float* __restrict__ dout) {
  int lane = threadIdx.x & 31, wave = threadIdx.x >> 5;
  int mt = blockIdx.x, b = blockIdx.y, nt = wave;
  const v16h* ap = (const v16h*)(apack + ((size_t)b * 256 + mt) * 8 * 512);
  const v16h* wp = (const v16h*)wpack;
  v8f acc = {};
#pragma unroll
  for (int kt = 0; kt < 8; ++kt) {
    if (kt < 7) __builtin_prefetch(&ap[(kt + 1) * 32 + lane], 0, 3);
    acc = wmma32(ap[kt * 32 + lane], wp[(kt * 8 + nt) * 32 + lane], acc);
  }
  int n = nt * 16 + (lane & 15);
  int hi = (lane >> 4) << 3;
  float bv = bias[n];
  size_t oi = ((size_t)b * 128 + n) * 4096 + mt * 16 + hi;
  float4 r0 = *(const float4*)(outb + oi);
  float4 r1 = *(const float4*)(outb + oi + 4);
  *(float4*)(dout + oi) = make_float4(r0.x + acc[0] + bv, r0.y + acc[1] + bv,
                                      r0.z + acc[2] + bv, r0.w + acc[3] + bv);
  *(float4*)(dout + oi + 4) = make_float4(r1.x + acc[4] + bv, r1.y + acc[5] + bv,
                                          r1.z + acc[6] + bv, r1.w + acc[7] + bv);
}

// ---------------------------------------------------------------------------
extern "C" void kernel_launch(void* const* d_in, const int* in_sizes, int n_in,
                              void* d_out, int out_size, void* d_ws,
                              size_t ws_size, hipStream_t stream) {
  (void)in_sizes; (void)n_in; (void)out_size; (void)ws_size;
  const float* x        = (const float*)d_in[0];
  const float* stokw    = (const float*)d_in[1];
  const float* stokb    = (const float*)d_in[2];
  const float* sdetw    = (const float*)d_in[3];
  const float* sdetb    = (const float*)d_in[4];
  const float* sm_in_w  = (const float*)d_in[5];
  const float* sm_in_b  = (const float*)d_in[6];
  const float* sm_cw    = (const float*)d_in[7];
  const float* sm_cb    = (const float*)d_in[8];
  const float* sm_Al    = (const float*)d_in[9];
  const float* sm_D     = (const float*)d_in[10];
  const float* sm_dtb   = (const float*)d_in[11];
  const float* sm_rms   = (const float*)d_in[12];
  const float* sm_ow    = (const float*)d_in[13];
  const float* cm_in_w  = (const float*)d_in[14];
  const float* cm_in_b  = (const float*)d_in[15];
  const float* cm_cw    = (const float*)d_in[16];
  const float* cm_cb    = (const float*)d_in[17];
  const float* cm_Al    = (const float*)d_in[18];
  const float* cm_D     = (const float*)d_in[19];
  const float* cm_dtb   = (const float*)d_in[20];
  const float* cm_rms   = (const float*)d_in[21];
  const float* cm_ow    = (const float*)d_in[22];
  const float* crossW   = (const float*)d_in[23];
  const float* n1w      = (const float*)d_in[24];
  const float* n1b      = (const float*)d_in[25];
  const float* n2w      = (const float*)d_in[26];
  const float* n2b      = (const float*)d_in[27];
  const float* fc1w     = (const float*)d_in[28];
  const float* fc1b     = (const float*)d_in[29];
  const float* dww      = (const float*)d_in[30];
  const float* dwb      = (const float*)d_in[31];
  const float* fc2w     = (const float*)d_in[32];
  const float* fc2b     = (const float*)d_in[33];
  const float* skip     = (const float*)d_in[34];
  float* dout = (float*)d_out;

  // workspace carve; whole pipeline is L2-resident on MI455X (192MB L2)
  float* ws   = (float*)d_ws;
  float* gf   = ws;                  // (B,L,8)
  float* gf2  = gf + 131072;         // (B,L,8)
  float* seed = gf2 + 131072;        // (B,C,L)
  float* xn   = seed + 2097152;      // (B,L,C)
  float* xsp  = xn + 2097152;        // (G,B,L,16)
  float* ysA  = xsp + 2097152;       // (G,B,L,16)
  float* ysB  = ysA + 2097152;       // (G,B,L,16)
  float* outb = ysB + 2097152;       // (B,C,L)
  float* h1   = outb + 2097152;      // (B,256,L)
  float* gm   = h1 + 4194304;        // (G,B,16)
  _Float16* apack1 = (_Float16*)(gm + 512);  // fc1 A frags: B*256*4*512
  _Float16* apack2 = apack1 + 2097152;       // fc2 A frags: B*256*8*512
  _Float16* wpk1   = apack2 + 4194304;       // fc1 W frags: 4*16*512
  _Float16* wpk2   = wpk1 + 32768;           // fc2 W frags: 8*8*512
  _Float16* mpk_sm = wpk2 + 32768;           // small mamba in_w frags: 5*512
  _Float16* mpk_c0 = mpk_sm + 2560;          // cascade0 in_w frags: 13*512
  _Float16* mpk_c1 = mpk_c0 + 6656;          // cascade1 in_w frags: 13*512

  // weight fragment packing (tiny, every call: deterministic)
  pack_w_kernel<<<128, 256, 0, stream>>>(fc1w, wpk1, 128, 16, 4, 128, 256, 32768);
  pack_w_kernel<<<128, 256, 0, stream>>>(fc2w, wpk2, 256, 8, 8, 256, 128, 32768);
  pack_w_kernel<<<10, 256, 0, stream>>>(sm_in_w, mpk_sm, 8, 5, 1, 8, 66, 2560);
  pack_w_kernel<<<26, 256, 0, stream>>>(cm_in_w, mpk_c0, 16, 13, 1, 16, 196, 6656);
  pack_w_kernel<<<26, 256, 0, stream>>>(cm_in_w + 196 * 16, mpk_c1, 16, 13, 1, 16, 196, 6656);

  seed_tok_kernel<<<512, 256, 0, stream>>>(x, stokw, stokb, gf);

  constexpr size_t ldsS = mamba_lds_floats<8, 66, 2, 16>() * sizeof(float);
  mamba_seq_kernel<8, 66, 2, 16><<<4, 256, ldsS, stream>>>(
      gf, gf2, gf, nullptr, mpk_sm, sm_in_b, sm_cw, sm_cb, sm_Al, sm_D,
      sm_dtb, sm_rms, sm_ow, 4096);

  seed_combine_kernel<<<8192, 256, 0, stream>>>(x, gf2, sdetw, sdetb, seed);
  norm1_kernel<<<16384, 128, 0, stream>>>(x, n1w, n1b, xn);
  split_kernel<<<8192, 256, 0, stream>>>(xn, xsp);

  constexpr size_t ldsC = mamba_lds_floats<16, 196, 4, 64>() * sizeof(float);
  mamba_seq_kernel<16, 196, 4, 64><<<32, 256, ldsC, stream>>>(
      xsp, ysA, nullptr, nullptr, mpk_c0, cm_in_b, cm_cw, cm_cb, cm_Al, cm_D,
      cm_dtb, cm_rms, cm_ow, 4096);
  mamba_seq_kernel<16, 196, 4, 64><<<32, 256, ldsC, stream>>>(
      ysA, ysB, xsp, skip, mpk_c1, cm_in_b + 196, cm_cw + 160 * 3,
      cm_cb + 160, cm_Al + 4, cm_D + 4, cm_dtb + 4, cm_rms + 32,
      cm_ow + 16 * 32, 4096);

  group_mean_kernel<<<32, 256, 0, stream>>>(ysB, gm);
  mix_norm2_kernel<<<16384, 128, 0, stream>>>(ysB, gm, crossW, seed, n2w, n2b,
                                              outb, apack1);

  fc1_wmma_kernel<<<dim3(256, 2, 4), 256, 0, stream>>>(apack1, wpk1, fc1b, h1);
  dwconv_gelu_kernel<<<16384, 256, 0, stream>>>(h1, dww, dwb, apack2);
  fc2_wmma_kernel<<<dim3(256, 4), 256, 0, stream>>>(apack2, wpk2, fc2b, outb, dout);
}
